// HyperLinearSelfAttention_54812372631771
// MI455X (gfx1250) — matrix operand order repaired and checked by reading, not compiled, not run on hardware
//
#include <hip/hip_runtime.h>

#define B_    8
#define NPTS  65536          // 256*256 points per batch
#define C_    64
#define H_    4
#define D_    16
#define ROWS  256            // rows per tile
#define CHUNK 4              // tiles per pass1 block
#define BLOCKS_X (NPTS/(ROWS*CHUNK))   // 64 pass1 blocks per batch
#define LN_EPSF 1e-5f

typedef __attribute__((ext_vector_type(16))) __bf16        v16bf;
typedef __attribute__((ext_vector_type(8)))  float         v8f;
typedef __attribute__((ext_vector_type(4)))  unsigned int  v4u;

union BF16Frag { v16bf v; v4u q[2]; unsigned int u32[8]; unsigned short u16[16]; };

__device__ __forceinline__ unsigned short f2bf(float f) {
  __bf16 h = (__bf16)f;                       // native cvt (RNE) where available
  return __builtin_bit_cast(unsigned short, h);
}
__device__ __forceinline__ float bf2f(unsigned short u) {
  return __uint_as_float(((unsigned int)u) << 16);
}

__device__ __forceinline__ v8f wmma_bf16(v16bf a, v16bf b, v8f c) {
  return __builtin_amdgcn_wmma_f32_16x16x32_bf16(false, a, false, b, (short)0, c,
                                                 false, false);
}

// 32-bit LDS byte address of a shared-memory object (for async-DMA VDST field)
__device__ __forceinline__ unsigned lds_addr_of(const void* p) {
  return (unsigned)(size_t)(__attribute__((address_space(3))) const unsigned char*)p;
}

// Issue one tile (ROWS*C_ fp32 = 64KB) of async global->LDS DMA: 16 x b128/lane.
__device__ __forceinline__ void async_tile_load(void* ldsDst, const float* gsrc, int t) {
  unsigned base = lds_addr_of(ldsDst);
#pragma unroll
  for (int i = 0; i < 16; ++i) {
    unsigned off = (unsigned)((t + i * 256) * 16);   // bytes
    asm volatile("global_load_async_to_lds_b128 %0, %1, %2"
                 :: "v"(base + off), "v"(off), "s"(gsrc)
                 : "memory");
  }
}

// ---- fragment helpers, wave32 layouts, contiguous k = 16*(lane/16) + element ----------------
// A (16x32 bf16) built from fp32 row-major LDS tile, converting on the fly.
__device__ __forceinline__ v16bf load_A_f32(const float* lds, int row0, int k0) {
  int lane = threadIdx.x & 31;
  int m  = lane & 15;
  int hi = lane >> 4;
  BF16Frag a;
#pragma unroll
  for (int v = 0; v < 8; ++v) {
    int kb = (hi << 4) + (v << 1);
    float x0 = lds[(row0 + m) * C_ + k0 + kb];
    float x1 = lds[(row0 + m) * C_ + k0 + kb + 1];
    a.u16[2 * v]     = f2bf(x0);
    a.u16[2 * v + 1] = f2bf(x1);
  }
  return a.v;
}

// Any fragment stored pre-swizzled: each lane's 16 bf16 are 32 contiguous bytes.
__device__ __forceinline__ v16bf load_frag16(const unsigned short* base) {
  BF16Frag f;
  const v4u* p = (const v4u*)base;
  f.q[0] = p[0];
  f.q[1] = p[1];
  return f.v;
}

__device__ __forceinline__ void store_C_f32(float* buf, int row0, int col0, int ldn, v8f acc) {
  int lane  = threadIdx.x & 31;
  int n     = lane & 15;
  int mbase = (lane >> 4) * 8;
#pragma unroll
  for (int i = 0; i < 8; ++i)
    buf[(row0 + mbase + i) * ldn + col0 + n] = acc[i];
}

// Scatter projection C-tile into A-swizzled K buffer: rl -> hi=rl>>4, j=rl&15
__device__ __forceinline__ void store_C_Ksw(unsigned short* sKsw, int wave, int h, int sub, v8f acc) {
  int lane  = threadIdx.x & 31;
  int n     = lane & 15;             // n == d
  int mbase = (lane >> 4) * 8;
#pragma unroll
  for (int i = 0; i < 8; ++i) {
    int rl  = sub * 16 + mbase + i;
    int hiK = rl >> 4;
    int jK  = rl & 15;
    sKsw[(((wave * H_ + h) * 32) + hiK * 16 + n) * 16 + jK] = f2bf(acc[i]);
  }
}
// Scatter projection C-tile into B-swizzled V buffer: rl -> hi=rl>>4, j=rl&15
__device__ __forceinline__ void store_C_Vsw(unsigned short* sVsw, int wave, int h, int sub, v8f acc) {
  int lane  = threadIdx.x & 31;
  int n     = lane & 15;             // n == e
  int mbase = (lane >> 4) * 8;
#pragma unroll
  for (int i = 0; i < 8; ++i) {
    int rl = sub * 16 + mbase + i;
    sVsw[(((wave * H_ + h) * 32) + (rl >> 4) * 16 + n) * 16 + (rl & 15)] = f2bf(acc[i]);
  }
}

// Scatter Q C-tile into A-swizzled layout for the out = Q @ Bbig GEMM.
__device__ __forceinline__ void store_C_Qsw(unsigned short* sQsw, int rt, int ct, v8f acc) {
  int lane   = threadIdx.x & 31;
  int n      = lane & 15;
  int mbase  = (lane >> 4) * 8;
  int klocal = (ct & 1) * 16 + n;          // k within the 32-wide kk block
  int hiA    = klocal >> 4;
  int jA     = klocal & 15;
  int kkidx  = ct >> 1;
#pragma unroll
  for (int i = 0; i < 8; ++i) {
    int m = mbase + i;
    sQsw[(((rt * 2 + kkidx) * 32) + hiA * 16 + m) * 16 + jA] = f2bf(acc[i]);
  }
}

__device__ __forceinline__ float gelu_tanh(float x) {
  const float k0 = 0.7978845608028654f, k1 = 0.044715f;
  return 0.5f * x * (1.0f + tanhf(k0 * (x + k1 * x * x * x)));
}

// ---------------------------------------------------------------------------
// Kernel A: hypernetwork -> bf16 weights in B-fragment order:
//   W[proj][b][ kkidx(2) ][ ct(4) ][ lane(32) ][ j(16) ]
// ---------------------------------------------------------------------------
__global__ __launch_bounds__(64) void hyper_kernel(
    const float* __restrict__ p,
    const float* __restrict__ qw1, const float* __restrict__ qb1,
    const float* __restrict__ qw2, const float* __restrict__ qb2,
    const float* __restrict__ kw1, const float* __restrict__ kb1,
    const float* __restrict__ kw2, const float* __restrict__ kb2,
    const float* __restrict__ vw1, const float* __restrict__ vb1,
    const float* __restrict__ vw2, const float* __restrict__ vb2,
    unsigned short* __restrict__ Wbf)
{
  int proj = blockIdx.x;                 // 0=q 1=k 2=v
  int b    = blockIdx.y;
  const float* w1 = (proj == 0) ? qw1 : (proj == 1) ? kw1 : vw1;
  const float* b1 = (proj == 0) ? qb1 : (proj == 1) ? kb1 : vb1;
  const float* w2 = (proj == 0) ? qw2 : (proj == 1) ? kw2 : vw2;
  const float* b2 = (proj == 0) ? qb2 : (proj == 1) ? kb2 : vb2;

  __shared__ float hid[64];
  int t = threadIdx.x;                   // t == output column hd
  float acc = b1[t];
#pragma unroll
  for (int i = 0; i < 16; ++i) acc += p[b * 16 + i] * w1[i * 64 + t];
  hid[t] = gelu_tanh(acc);
  __syncthreads();

  unsigned short* out = Wbf + (size_t)(proj * B_ + b) * (C_ * C_);
  int n  = t & 15;
  int ct = t >> 4;
  for (int c = 0; c < C_; ++c) {
    float a = b2[c * 64 + t];
#pragma unroll 16
    for (int j = 0; j < 64; ++j) a += hid[j] * w2[(size_t)j * 4096 + c * 64 + t];
    int kkidx = c >> 5;
    int hi    = (c >> 4) & 1;
    int jj    = c & 15;
    out[(((kkidx * 4 + ct) * 32) + hi * 16 + n) * 16 + jj] = f2bf(a);
  }
}

// ---------------------------------------------------------------------------
// Kernel B: K/V projection + LN + K^T V, async double-buffered X staging
// ---------------------------------------------------------------------------
__global__ __launch_bounds__(256) void pass1_kernel(
    const float* __restrict__ X, const unsigned short* __restrict__ Wbf,
    const float* __restrict__ lng, const float* __restrict__ lnb,
    float* __restrict__ part)
{
  __shared__ __align__(16) float sXf[2][ROWS * C_];               // 128 KB, fp32 X tiles
  __shared__ __align__(16) unsigned short sKsw[8 * H_ * 32 * 16]; // 32 KB, A-swizzled K
  __shared__ __align__(16) unsigned short sVsw[8 * H_ * 32 * 16]; // 32 KB, B-swizzled V
  __shared__ float sG[C_];
  __shared__ float sBv[C_];
  float* sPart = (float*)sXf;  // 32 KB needed, aliases dead X buffer after the loop

  int t    = threadIdx.x;
  int wave = t >> 5;
  int lane = t & 31;
  int b    = blockIdx.y;

  if (t < C_) { sG[t] = lng[t]; sBv[t] = lnb[t]; }   // lnk params for BOTH K and V

  const float* Xb = X + (size_t)b * NPTS * C_;

  // kick off chunk 0 DMA before touching anything else
  async_tile_load(&sXf[0][0], Xb + (size_t)(blockIdx.x * CHUNK + 0) * ROWS * C_, t);

  // Hoist all W fragments into registers (2x global_load_b128 each, coalesced).
  const unsigned short* wkG = Wbf + (size_t)(1 * B_ + b) * (C_ * C_);
  const unsigned short* wvG = Wbf + (size_t)(2 * B_ + b) * (C_ * C_);
  v16bf fWk[8], fWv[8];
#pragma unroll
  for (int f = 0; f < 8; ++f) {          // f = kkidx*4 + ct
    fWk[f] = load_frag16(wkG + (f * 32 + lane) * 16);
    fWv[f] = load_frag16(wvG + (f * 32 + lane) * 16);
  }

  v8f accKV[H_] = {};

  for (int ch = 0; ch < CHUNK; ++ch) {
    int cur = ch & 1;
    if (ch + 1 < CHUNK) {
      // issue next tile's DMA, then wait until only those 16 remain outstanding
      async_tile_load(&sXf[cur ^ 1][0],
                      Xb + (size_t)(blockIdx.x * CHUNK + ch + 1) * ROWS * C_, t);
      asm volatile("s_wait_asynccnt 16" ::: "memory");
    } else {
      asm volatile("s_wait_asynccnt 0" ::: "memory");
    }
    __syncthreads();
    const float* sXc = &sXf[cur][0];

    // projections: K = X @ Wk, V = X @ Wv; ct == head h
#pragma unroll
    for (int sub = 0; sub < 2; ++sub) {
      int row0 = (wave * 2 + sub) * 16;
      v16bf a0 = load_A_f32(sXc, row0, 0);
      v16bf a1 = load_A_f32(sXc, row0, 32);
#pragma unroll
      for (int ct = 0; ct < 4; ++ct) {
        v8f ck = {}; v8f cv = {};
        ck = wmma_bf16(a0, fWk[ct],     ck);
        ck = wmma_bf16(a1, fWk[4 + ct], ck);
        cv = wmma_bf16(a0, fWv[ct],     cv);
        cv = wmma_bf16(a1, fWv[4 + ct], cv);
        store_C_Ksw(sKsw, wave, ct, sub, ck);
        store_C_Vsw(sVsw, wave, ct, sub, cv);
      }
    }
    __syncthreads();

    // LayerNorm over all 64 channels; thread owns V row rl and K row rl with bits 3,4 swapped, of its wave.
    {
      int rl  = lane;
      int hiK = (rl >> 3) & 1;
      int jK  = (rl & 7) | ((rl >> 1) & 8);
      int hiV = rl >> 4;
      int jV  = rl & 15;
      unsigned short* bK = sKsw + (wave * H_) * 32 * 16;
      unsigned short* bV = sVsw + (wave * H_) * 32 * 16;
      float xv[C_];

      float s = 0.f, ss = 0.f;
#pragma unroll
      for (int h = 0; h < H_; ++h)
#pragma unroll
        for (int d = 0; d < 16; ++d) {
          float x = bf2f(bK[((h * 32) + hiK * 16 + d) * 16 + jK]);
          xv[h * 16 + d] = x; s += x; ss += x * x;
        }
      float mu = s * (1.f / C_);
      float r  = rsqrtf(ss * (1.f / C_) - mu * mu + LN_EPSF);
#pragma unroll
      for (int h = 0; h < H_; ++h)
#pragma unroll
        for (int d = 0; d < 16; ++d) {
          int c = h * 16 + d;
          bK[((h * 32) + hiK * 16 + d) * 16 + jK] = f2bf((xv[c] - mu) * r * sG[c] + sBv[c]);
        }

      s = 0.f; ss = 0.f;
#pragma unroll
      for (int h = 0; h < H_; ++h)
#pragma unroll
        for (int e = 0; e < 16; ++e) {
          float x = bf2f(bV[((h * 32) + hiV * 16 + e) * 16 + jV]);
          xv[h * 16 + e] = x; s += x; ss += x * x;
        }
      mu = s * (1.f / C_);
      r  = rsqrtf(ss * (1.f / C_) - mu * mu + LN_EPSF);
#pragma unroll
      for (int h = 0; h < H_; ++h)
#pragma unroll
        for (int e = 0; e < 16; ++e) {
          int c = h * 16 + e;
          bV[((h * 32) + hiV * 16 + e) * 16 + jV] = f2bf((xv[c] - mu) * r * sG[c] + sBv[c]);
        }
    }
    __syncthreads();

    // K^T V per head: pre-swizzled operands -> 2x ds_load_b128 each.
#pragma unroll
    for (int h = 0; h < H_; ++h) {
      v16bf a  = load_frag16(sKsw + ((wave * H_ + h) * 32 + lane) * 16);
      v16bf bv = load_frag16(sVsw + ((wave * H_ + h) * 32 + lane) * 16);
      accKV[h] = wmma_bf16(a, bv, accKV[h]);
    }
    __syncthreads();
  }

  // per-wave partials -> LDS, fixed-order block reduction -> global partials
#pragma unroll
  for (int h = 0; h < H_; ++h)
    store_C_f32(sPart + (wave * H_ + h) * 256, 0, 0, 16, accKV[h]);
  __syncthreads();

  float* pg = part + ((size_t)b * BLOCKS_X + blockIdx.x) * (H_ * 256);
#pragma unroll
  for (int i = 0; i < 4; ++i) {
    int idx = t + i * 256;
    float s = 0.f;
#pragma unroll
    for (int w = 0; w < 8; ++w) s += sPart[w * (H_ * 256) + idx];
    pg[idx] = s;
  }
}

// ---------------------------------------------------------------------------
// Kernel R: deterministic reduction, then emit per-batch 64x64 block-diagonal
// Bbig[k=h*16+d][o=e*4+h] = KV[h][d][e]/N in bf16 B-fragment order.
// ---------------------------------------------------------------------------
__global__ __launch_bounds__(1024) void reduce_kv(const float* __restrict__ part,
                                                  unsigned short* __restrict__ Bbig)
{
  __shared__ float sKV[1024];
  int b = blockIdx.x;
  int t = threadIdx.x;            // flat (h,d,e)
  float s = 0.f;
  for (int blk = 0; blk < BLOCKS_X; ++blk)
    s += part[((size_t)b * BLOCKS_X + blk) * 1024 + t];
  sKV[t] = s;
  __syncthreads();

  unsigned short* dst = Bbig + (size_t)b * (C_ * C_);
  const float invN = 1.0f / (float)NPTS;
#pragma unroll
  for (int i = 0; i < 4; ++i) {
    int f    = t + i * 1024;      // flat B-fragment index: ((frag*32+lane)*16+j)
    int jB   = f & 15;
    int r    = f >> 4;
    int lane = r & 31;
    int frag = r >> 5;            // kkidx*4 + ct
    int ct   = frag & 3;
    int kk   = frag >> 2;
    int k    = kk * 32 + (lane >> 4) * 16 + jB;   // input channel h*16+d
    int o    = ct * 16 + (lane & 15);             // output channel e*4+h
    int hk   = k >> 4, d = k & 15;
    int ho   = o & 3,  e = o >> 2;
    float val = (hk == ho) ? sKV[(hk * 16 + d) * 16 + e] * invN : 0.f;
    dst[f] = f2bf(val);
  }
}

// ---------------------------------------------------------------------------
// Kernel C: Q projection (WMMA) + out = Q @ Bbig (WMMA), async X staging
// ---------------------------------------------------------------------------
__global__ __launch_bounds__(256) void pass2_kernel(
    const float* __restrict__ X, const unsigned short* __restrict__ Wbf,
    const unsigned short* __restrict__ Bbig, float* __restrict__ out)
{
  __shared__ __align__(16) float sXf[ROWS * C_];                    // 64 KB
  __shared__ __align__(16) unsigned short sQsw[16 * 2 * 32 * 16];   // 32 KB, A-swizzled Q

  int t    = threadIdx.x;
  int wave = t >> 5;
  int lane = t & 31;
  int b    = blockIdx.y;
  int tile = blockIdx.x;

  async_tile_load(&sXf[0],
                  X + ((size_t)b * NPTS + (size_t)tile * ROWS) * C_, t);

  const unsigned short* wqG = Wbf + (size_t)(0 * B_ + b) * (C_ * C_);
  const unsigned short* bbG = Bbig + (size_t)b * (C_ * C_);
  v16bf fWq[8], fB[8];
#pragma unroll
  for (int f = 0; f < 8; ++f) {
    fWq[f] = load_frag16(wqG + (f * 32 + lane) * 16);
    fB[f]  = load_frag16(bbG + (f * 32 + lane) * 16);
  }

  asm volatile("s_wait_asynccnt 0" ::: "memory");
  __syncthreads();

#pragma unroll
  for (int sub = 0; sub < 2; ++sub) {
    int rt   = wave * 2 + sub;
    int row0 = rt * 16;
    v16bf a0 = load_A_f32(sXf, row0, 0);
    v16bf a1 = load_A_f32(sXf, row0, 32);
#pragma unroll
    for (int ct = 0; ct < 4; ++ct) {
      v8f cq = {};
      cq = wmma_bf16(a0, fWq[ct],     cq);
      cq = wmma_bf16(a1, fWq[4 + ct], cq);
      store_C_Qsw(sQsw, rt, ct, cq);
    }
  }
  __syncthreads();

  // out = Q @ Bbig  (division by N and channel permutation folded into Bbig)
  {
    int n     = lane & 15;
    int mbase = (lane >> 4) * 8;
#pragma unroll
    for (int sub = 0; sub < 2; ++sub) {
      int rt = wave * 2 + sub;
      v16bf q0 = load_frag16(sQsw + ((rt * 2 + 0) * 32 + lane) * 16);
      v16bf q1 = load_frag16(sQsw + ((rt * 2 + 1) * 32 + lane) * 16);
      float* og = out + ((size_t)b * NPTS + (size_t)tile * ROWS + rt * 16) * C_;
#pragma unroll
      for (int ct = 0; ct < 4; ++ct) {
        v8f o = {};
        o = wmma_bf16(q0, fB[ct],     o);
        o = wmma_bf16(q1, fB[4 + ct], o);
#pragma unroll
        for (int i = 0; i < 8; ++i)
          og[(mbase + i) * C_ + ct * 16 + n] = o[i];
      }
    }
  }
}

// ---------------------------------------------------------------------------
extern "C" void kernel_launch(void* const* d_in, const int* in_sizes, int n_in,
                              void* d_out, int out_size, void* d_ws, size_t ws_size,
                              hipStream_t stream) {
  const float* X    = (const float*)d_in[0];
  const float* p    = (const float*)d_in[1];
  const float* qw1  = (const float*)d_in[2];
  const float* qb1  = (const float*)d_in[3];
  const float* qw2  = (const float*)d_in[4];
  const float* qb2  = (const float*)d_in[5];
  const float* kw1  = (const float*)d_in[6];
  const float* kb1  = (const float*)d_in[7];
  const float* kw2  = (const float*)d_in[8];
  const float* kb2  = (const float*)d_in[9];
  const float* vw1  = (const float*)d_in[10];
  const float* vb1  = (const float*)d_in[11];
  const float* vw2  = (const float*)d_in[12];
  const float* vb2  = (const float*)d_in[13];
  const float* lnkg = (const float*)d_in[14];
  const float* lnkb = (const float*)d_in[15];
  // d_in[16], d_in[17] (lnv_g/lnv_b) intentionally unused: reference applies lnk to V too.

  // workspace layout
  unsigned short* Wbf = (unsigned short*)d_ws;                       // 196608 B
  size_t off = (size_t)3 * B_ * C_ * C_ * sizeof(unsigned short);
  float* part = (float*)((char*)d_ws + off);                         // 2 MB
  off += (size_t)B_ * BLOCKS_X * 1024 * sizeof(float);
  unsigned short* Bbig = (unsigned short*)((char*)d_ws + off);       // 64 KB

  hyper_kernel<<<dim3(3, B_), 64, 0, stream>>>(p, qw1, qb1, qw2, qb2,
                                               kw1, kb1, kw2, kb2,
                                               vw1, vb1, vw2, vb2, Wbf);
  pass1_kernel<<<dim3(BLOCKS_X, B_), 256, 0, stream>>>(X, Wbf, lnkg, lnkb, part);
  reduce_kv<<<dim3(B_), 1024, 0, stream>>>(part, Bbig);
  pass2_kernel<<<dim3(NPTS / ROWS, B_), 256, 0, stream>>>(X, Wbf, Bbig, (float*)d_out);
}
